// CRvNN_31258771981067
// MI455X (gfx1250) — compile-verified, matrix operand order checked by reading, not run on hardware
//
#include <hip/hip_runtime.h>
#include <hip/hip_bf16.h>
#include <math.h>

#define NB   64
#define SCT  128
#define S    130
#define D    1024
#define NS   (NB * S)     // 8320 rows, 260 tiles of 32
#define STEPS 25

typedef __attribute__((ext_vector_type(16))) __bf16       bf16x16;
typedef __attribute__((ext_vector_type(8)))  float        f32x8;
typedef __attribute__((ext_vector_type(4)))  float        f32x4;
typedef __attribute__((ext_vector_type(4)))  unsigned int u32x4;
typedef unsigned short u16;

union FragAB { bf16x16 bf; u32x4 u[2]; };

__device__ __forceinline__ u16 f2bf(float x) {
    unsigned int u = __float_as_uint(x);
    unsigned int r = u + 0x7FFFu + ((u >> 16) & 1u);   // RNE
    if ((u & 0x7F800000u) == 0x7F800000u) r = u;       // inf/nan passthrough
    return (u16)(r >> 16);
}
__device__ __forceinline__ float gelu_f(float x) {
    return 0.5f * x * (1.0f + erff(x * 0.7071067811865476f));
}
__device__ __forceinline__ float sigm(float x) { return 1.0f / (1.0f + __expf(-x)); }
__device__ __forceinline__ f32x4 sig4(f32x4 x) {
    f32x4 r; r[0]=sigm(x[0]); r[1]=sigm(x[1]); r[2]=sigm(x[2]); r[3]=sigm(x[3]); return r;
}
__device__ __forceinline__ f32x4 splat4(float x) { f32x4 v = {x,x,x,x}; return v; }

// ---------------------------------------------------------------------------
// Generic bf16 WMMA GEMM: C[M,N] = act(A[M,K] * Bt[N,K]^T + bias)
// A row-major (lda elems), Bt row-major N x K (ldb elems).
// Wave tile: 32M x 64N, double-buffered K pipeline (K/32 chunks, always even).
// ---------------------------------------------------------------------------
__global__ __launch_bounds__(256)
void k_gemm(const u16* __restrict__ A, int lda,
            const u16* __restrict__ Bt, int ldb,
            float* __restrict__ C, int ldc,
            const float* __restrict__ bias,
            int tilesM, int tilesN, int K, int act)
{
    int wid = (blockIdx.x << 3) + (threadIdx.x >> 5);
    if (wid >= tilesM * tilesN) return;          // wave-uniform exit
    int tm = wid / tilesN;
    int tn = wid - tm * tilesN;
    int lane = threadIdx.x & 31;
    int half = lane >> 4;        // 0: K 0-7/16-23 ; 1: K 8-15/24-31
    int r    = lane & 15;

    const u16* arow0 = A  + (size_t)(tm * 32 + r)      * (size_t)lda + half * 8;
    const u16* arow1 = arow0 + (size_t)16 * (size_t)lda;
    const u16* brow  = Bt + (size_t)(tn * 64 + r)      * (size_t)ldb + half * 8;
    const size_t bstep = (size_t)16 * (size_t)ldb;

    f32x8 acc[8] = {};
    FragAB a0c, a1c, b0c, b1c, b2c, b3c;   // current
    FragAB a0n, a1n, b0n, b1n, b2n, b3n;   // next

    auto LD1 = [&](FragAB& f, const u16* p) {
        f.u[0] = *(const u32x4*)p;
        f.u[1] = *(const u32x4*)(p + 16);
    };
    auto LD6 = [&](FragAB& a0, FragAB& a1, FragAB& b0, FragAB& b1,
                   FragAB& b2, FragAB& b3, int k) {
        LD1(a0, arow0 + k); LD1(a1, arow1 + k);
        LD1(b0, brow + k);            LD1(b1, brow + bstep + k);
        LD1(b2, brow + 2 * bstep + k); LD1(b3, brow + 3 * bstep + k);
        __builtin_prefetch(arow0 + k + 512, 0, 3);
    };
    auto MM = [&](FragAB& a0, FragAB& a1, FragAB& b0, FragAB& b1,
                  FragAB& b2, FragAB& b3) {
        acc[0] = __builtin_amdgcn_wmma_f32_16x16x32_bf16(false, a0.bf, false, b0.bf, (short)0, acc[0], false, false);
        acc[1] = __builtin_amdgcn_wmma_f32_16x16x32_bf16(false, a0.bf, false, b1.bf, (short)0, acc[1], false, false);
        acc[2] = __builtin_amdgcn_wmma_f32_16x16x32_bf16(false, a0.bf, false, b2.bf, (short)0, acc[2], false, false);
        acc[3] = __builtin_amdgcn_wmma_f32_16x16x32_bf16(false, a0.bf, false, b3.bf, (short)0, acc[3], false, false);
        acc[4] = __builtin_amdgcn_wmma_f32_16x16x32_bf16(false, a1.bf, false, b0.bf, (short)0, acc[4], false, false);
        acc[5] = __builtin_amdgcn_wmma_f32_16x16x32_bf16(false, a1.bf, false, b1.bf, (short)0, acc[5], false, false);
        acc[6] = __builtin_amdgcn_wmma_f32_16x16x32_bf16(false, a1.bf, false, b2.bf, (short)0, acc[6], false, false);
        acc[7] = __builtin_amdgcn_wmma_f32_16x16x32_bf16(false, a1.bf, false, b3.bf, (short)0, acc[7], false, false);
    };

    // software pipeline over K chunks (nc = K/32, always even for our shapes)
    int nc = K >> 5;
    LD6(a0c, a1c, b0c, b1c, b2c, b3c, 0);
    int k = 32;
    for (int j = 0; j < (nc >> 1) - 1; ++j) {
        LD6(a0n, a1n, b0n, b1n, b2n, b3n, k); k += 32;
        MM(a0c, a1c, b0c, b1c, b2c, b3c);
        LD6(a0c, a1c, b0c, b1c, b2c, b3c, k); k += 32;
        MM(a0n, a1n, b0n, b1n, b2n, b3n);
    }
    LD6(a0n, a1n, b0n, b1n, b2n, b3n, k);
    MM(a0c, a1c, b0c, b1c, b2c, b3c);
    MM(a0n, a1n, b0n, b1n, b2n, b3n);

#pragma unroll
    for (int s = 0; s < 2; ++s) {
#pragma unroll
        for (int t = 0; t < 4; ++t) {
            int col = tn * 64 + t * 16 + r;
            float bs = bias ? bias[col] : 0.0f;
#pragma unroll
            for (int q = 0; q < 8; ++q) {
                float v = acc[s * 4 + t][q] + bs;
                if (act) v = gelu_f(v);
                C[(size_t)(tm * 32 + s * 16 + half * 8 + q) * (size_t)ldc + col] = v;
            }
        }
    }
}

// ---------------------------------------------------------------------------
// Weight transpose + f32->bf16:  W[K,N] -> Wt[N,K]
// ---------------------------------------------------------------------------
__global__ void k_wt(const float* __restrict__ W, u16* __restrict__ Wt, int K, int N)
{
    long long idx = (long long)blockIdx.x * blockDim.x + threadIdx.x;
    if (idx >= (long long)K * N) return;
    int k  = (int)(idx % K);
    int nI = (int)(idx / K);
    Wt[(size_t)nI * K + k] = f2bf(W[(size_t)k * N + nI]);
}

// ---------------------------------------------------------------------------
// Row-wise f32->bf16 convert with optional per-row scale
// ---------------------------------------------------------------------------
__global__ void k_cvtbf(const float* __restrict__ src, unsigned long long srcStride,
                        u16* __restrict__ dst, unsigned long long dstStride,
                        int rowLen, const float* __restrict__ rowScale)
{
    int row = blockIdx.x;
    float sc = rowScale ? rowScale[row] : 1.0f;
    const float* s = src + (size_t)row * srcStride;
    u16* d = dst + (size_t)row * dstStride;
    for (int c = threadIdx.x; c < rowLen; c += blockDim.x)
        d[c] = f2bf(s[c] * sc);
}

// ---------------------------------------------------------------------------
// Augment: build seq (n,s,d) and mask/end/sel/last (n,s)
// ---------------------------------------------------------------------------
__global__ void k_augment(const float* __restrict__ seqin, const float* __restrict__ imask,
                          const float* __restrict__ STARTv, const float* __restrict__ ENDv,
                          float* __restrict__ seq, float* __restrict__ mask,
                          float* __restrict__ endm, float* __restrict__ selm,
                          float* __restrict__ lastm)
{
    int row = blockIdx.x;
    int n = row / S;
    int i = row - n * S;
    auto mye = [&](int ii) -> float {          // m_yes_end
        if (ii <= 1) return 1.0f;
        return imask[n * SCT + (ii - 2)];
    };
    auto mne = [&](int ii) -> float {          // m_no_end
        if (ii == 0) return 1.0f;
        if (ii <= SCT) return imask[n * SCT + (ii - 1)];
        return 0.0f;
    };
    float myei = mye(i), mnei = mne(i);
    float e    = myei - mnei;                              // end_mask
    float last = (i < S - 1) ? (mye(i + 1) - mne(i + 1)) : 0.0f;
    float mns  = (i == 0) ? 0.0f : myei;                   // m_no_start
    float sel  = mns * mnei * (1.0f - last);
    if (threadIdx.x == 0) {
        mask[row] = myei; endm[row] = e; selm[row] = sel; lastm[row] = last;
    }
    int c0 = threadIdx.x * 4;
    f32x4 base;
    if (i == 0)        base = *(const f32x4*)(STARTv + c0);
    else if (i <= SCT) base = *(const f32x4*)(seqin + ((size_t)n * SCT + (i - 1)) * D + c0);
    else               base = splat4(0.0f);
    f32x4 ev = *(const f32x4*)(ENDv + c0);
    f32x4 o  = e * ev + (1.0f - e) * base;
    *(f32x4*)(seq + (size_t)row * D + c0) = o;
}

// ---------------------------------------------------------------------------
// LayerNorm (init): seq = LN(src)*g+b, * mask
// ---------------------------------------------------------------------------
__global__ void k_ln1(const float* __restrict__ src, float* __restrict__ dst,
                      const float* __restrict__ g, const float* __restrict__ b,
                      const float* __restrict__ mask)
{
    __shared__ float rs[256], rq[256];
    int row = blockIdx.x, tid = threadIdx.x;
    int c0 = tid * 4;
    f32x4 v = *(const f32x4*)(src + (size_t)row * D + c0);
    float s1 = v[0] + v[1] + v[2] + v[3];
    float s2 = v[0]*v[0] + v[1]*v[1] + v[2]*v[2] + v[3]*v[3];
    rs[tid] = s1; rq[tid] = s2; __syncthreads();
    for (int off = 128; off > 0; off >>= 1) {
        if (tid < off) { rs[tid] += rs[tid + off]; rq[tid] += rq[tid + off]; }
        __syncthreads();
    }
    float mu  = rs[0] * (1.0f / D);
    float var = rq[0] * (1.0f / D) - mu * mu;
    float inv = rsqrtf(var + 1e-5f);
    float mk  = mask[row];
    f32x4 gv = *(const f32x4*)(g + c0);
    f32x4 bv = *(const f32x4*)(b + c0);
    f32x4 o  = ((v - splat4(mu)) * inv) * gv + bv;
    *(f32x4*)(dst + (size_t)row * D + c0) = o * mk;
}

__global__ void k_initstate(const float* __restrict__ mask, float* __restrict__ ep,
                            float* __restrict__ um)
{
    int i = blockIdx.x * blockDim.x + threadIdx.x;
    if (i < NS) ep[i] = mask[i];
    if (i < NB) um[i] = 1.0f;
}

__global__ void k_zero1(float* p) { if (threadIdx.x == 0 && blockIdx.x == 0) p[0] = 0.0f; }

// ---------------------------------------------------------------------------
// Neighbor probability matrices L, R  (one thread per (n,i,dir))
// ---------------------------------------------------------------------------
__global__ void k_neighbors(const float* __restrict__ ep, const float* __restrict__ mask,
                            float* __restrict__ Lm, float* __restrict__ Rm)
{
    int idx = blockIdx.x * blockDim.x + threadIdx.x;
    if (idx >= 2 * NS) return;
    int dir  = idx / NS;
    int rest = idx - dir * NS;
    int n = rest / S;
    int i = rest - n * S;
    const float* epn = ep + (size_t)n * S;
    const float* mk  = mask + (size_t)n * S;
    if (dir == 1) {
        float cs = 0.0f;
        float* out = Rm + ((size_t)n * S + i) * S;
        for (int j = 0; j < S; ++j) {
            float mat = (j > i) ? epn[j] * mk[j] : 0.0f;
            float prev = cs; cs += mat;
            float rem = fmaxf(1.0f - prev, 0.0f);
            out[j] = ((cs > 1.0f) ? rem : mat) * mk[j];
        }
    } else {
        int rr = S - 1 - i;
        float cs = 0.0f;
        float* out = Lm + ((size_t)n * S + i) * S;
        for (int jp = 0; jp < S; ++jp) {
            int o = S - 1 - jp;
            float mat = (jp > rr) ? epn[o] * mk[o] : 0.0f;
            float prev = cs; cs += mat;
            float rem = fmaxf(1.0f - prev, 0.0f);
            out[o] = ((cs > 1.0f) ? rem : mat) * mk[o];
        }
    }
}

// ---------------------------------------------------------------------------
// out[row,:] = sum_j M[row,j] * X[n,j,:]  (rows of M are ~2-sparse)
// writes f32 (optional) + bf16 slice of win
// ---------------------------------------------------------------------------
__global__ void k_rowmix(const float* __restrict__ Mmat, const float* __restrict__ X,
                         float* __restrict__ outF, u16* __restrict__ outBF, int bfOff)
{
    int row = blockIdx.x;
    int n = row / S;
    const float* Mrow = Mmat + (size_t)row * S;
    const float* Xn   = X + (size_t)n * S * D;
    int c0 = threadIdx.x * 4;
    f32x4 acc = splat4(0.0f);
    for (int j = 0; j < S; ++j) {
        float w = Mrow[j];
        if (w != 0.0f) acc += w * (*(const f32x4*)(Xn + (size_t)j * D + c0));
    }
    if (outF) *(f32x4*)(outF + (size_t)row * D + c0) = acc;
    u16* d = outBF + (size_t)row * (5 * D) + bfOff + c0;
    d[0] = f2bf(acc[0]); d[1] = f2bf(acc[1]); d[2] = f2bf(acc[2]); d[3] = f2bf(acc[3]);
}

// ---------------------------------------------------------------------------
// Scorer: per-row dot(h, W_scorer)+b ; global max (>=0 since zero column)
// ---------------------------------------------------------------------------
__global__ void k_scorer1(const float* __restrict__ h, const float* __restrict__ Ws,
                          const float* __restrict__ bs, float* __restrict__ sc,
                          float* __restrict__ mx)
{
    int wid = threadIdx.x >> 5, lane = threadIdx.x & 31;
    int row = blockIdx.x * 8 + wid;
    const float* hr = h + (size_t)row * D;
    float a = 0.0f;
    for (int c = lane; c < D; c += 32) a += hr[c] * Ws[c];
    for (int off = 16; off > 0; off >>= 1) a += __shfl_xor(a, off, 32);
    if (lane == 0) {
        float v = a + bs[0];
        sc[row] = v;
        if (v > 0.0f) atomicMax((int*)mx, __float_as_int(v));  // monotone for positives
    }
}

__global__ void k_scorer2(const float* __restrict__ sc, const float* __restrict__ sel,
                          const float* __restrict__ um, const float* __restrict__ mx,
                          float* __restrict__ tp)
{
    int idx = blockIdx.x * blockDim.x + threadIdx.x;
    if (idx >= NS) return;
    int n = idx / S;
    float m  = mx[0];
    float et = __expf(sc[idx] - m) * sel[idx];
    float en = __expf(-m);
    tp[idx] = et / (et + en + 2e-8f) * um[n];
}

// ---------------------------------------------------------------------------
// Gates + LN2 + state blend:  seq = tp*LN(g0*l1+g1*sq+g2*c3) + (1-tp)*sq
// ---------------------------------------------------------------------------
__global__ void k_comp(const float* __restrict__ contents, const float* __restrict__ l1,
                       float* __restrict__ seq, const float* __restrict__ tp,
                       const float* __restrict__ g, const float* __restrict__ b)
{
    __shared__ float rs[256], rq[256];
    int row = blockIdx.x, tid = threadIdx.x;
    int c0 = tid * 4;
    const float* crow = contents + (size_t)row * 4 * D;
    f32x4 g0 = sig4(*(const f32x4*)(crow + 0 * D + c0));
    f32x4 g1 = sig4(*(const f32x4*)(crow + 1 * D + c0));
    f32x4 g2 = sig4(*(const f32x4*)(crow + 2 * D + c0));
    f32x4 c3 = *(const f32x4*)(crow + 3 * D + c0);
    f32x4 l1v = *(const f32x4*)(l1 + (size_t)row * D + c0);
    f32x4 sqv = *(const f32x4*)(seq + (size_t)row * D + c0);
    f32x4 pre = g0 * l1v + g1 * sqv + g2 * c3;
    float s1 = pre[0] + pre[1] + pre[2] + pre[3];
    float s2 = pre[0]*pre[0] + pre[1]*pre[1] + pre[2]*pre[2] + pre[3]*pre[3];
    rs[tid] = s1; rq[tid] = s2; __syncthreads();
    for (int off = 128; off > 0; off >>= 1) {
        if (tid < off) { rs[tid] += rs[tid + off]; rq[tid] += rq[tid + off]; }
        __syncthreads();
    }
    float mu  = rs[0] * (1.0f / D);
    float var = rq[0] * (1.0f / D) - mu * mu;
    float inv = rsqrtf(var + 1e-5f);
    f32x4 gv = *(const f32x4*)(g + c0);
    f32x4 bv = *(const f32x4*)(b + c0);
    f32x4 comp = ((pre - splat4(mu)) * inv) * gv + bv;
    float t = tp[row];
    f32x4 ns = t * comp + (1.0f - t) * sqv;
    *(f32x4*)(seq + (size_t)row * D + c0) = ns;
}

// ---------------------------------------------------------------------------
// ep = ep*(1 - R@tp); um *= (sum(ep*sel) >= STOP)
// ---------------------------------------------------------------------------
__global__ void k_ep(const float* __restrict__ Rm, const float* __restrict__ tp,
                     const float* __restrict__ sel, float* __restrict__ ep,
                     float* __restrict__ um)
{
    __shared__ float red[256];
    int n = blockIdx.x, tid = threadIdx.x;
    float contrib = 0.0f;
    if (tid < S) {
        const float* Rrow = Rm + ((size_t)n * S + tid) * S;
        const float* tpn  = tp + (size_t)n * S;
        float a = 0.0f;
        for (int j = 0; j < S; ++j) a += Rrow[j] * tpn[j];
        float ne = ep[n * S + tid] * (1.0f - a);
        ep[n * S + tid] = ne;
        contrib = ne * sel[n * S + tid];
    }
    red[tid] = contrib; __syncthreads();
    for (int off = 128; off > 0; off >>= 1) {
        if (tid < off) red[tid] += red[tid + off];
        __syncthreads();
    }
    if (tid == 0) { if (red[0] < 0.1f) um[n] = 0.0f; }
}

// ---------------------------------------------------------------------------
// Outputs
// ---------------------------------------------------------------------------
__global__ void k_outcopy(const float* __restrict__ seq, const float* __restrict__ mask,
                          float* __restrict__ out)
{
    int blk = blockIdx.x;
    int n = blk / SCT, ii = blk - n * SCT, i = ii + 1;
    int c0 = threadIdx.x * 4;
    float mk = mask[n * S + i];
    f32x4 v = *(const f32x4*)(seq + ((size_t)n * S + i) * D + c0);
    *(f32x4*)(out + ((size_t)n * SCT + ii) * D + c0) = v * mk;
}

__global__ void k_outgs(const float* __restrict__ seq, const float* __restrict__ mask,
                        const float* __restrict__ lastm, float* __restrict__ out)
{
    int n = blockIdx.x;
    int c0 = threadIdx.x * 4;
    f32x4 acc = splat4(0.0f);
    for (int i = 0; i < S; ++i) {
        float lm = lastm[n * S + i];
        if (lm != 0.0f) {
            float mk = mask[n * S + i];
            acc += (lm * mk) * (*(const f32x4*)(seq + ((size_t)n * S + i) * D + c0));
        }
    }
    *(f32x4*)(out + (size_t)n * D + c0) = acc;
}

// ===========================================================================
extern "C" void kernel_launch(void* const* d_in, const int* in_sizes, int n_in,
                              void* d_out, int out_size, void* d_ws, size_t ws_size,
                              hipStream_t stream)
{
    const float* in_seq   = (const float*)d_in[0];
    const float* in_mask  = (const float*)d_in[1];
    const float* STARTv   = (const float*)d_in[2];
    const float* ENDv     = (const float*)d_in[3];
    const float* W_conv   = (const float*)d_in[4];
    const float* b_conv   = (const float*)d_in[5];
    const float* W_scorer = (const float*)d_in[6];
    const float* b_scorer = (const float*)d_in[7];
    const float* W_init   = (const float*)d_in[8];
    const float* b_init   = (const float*)d_in[9];
    const float* W_cell1  = (const float*)d_in[10];
    const float* b_cell1  = (const float*)d_in[11];
    const float* W_cell2  = (const float*)d_in[12];
    const float* b_cell2  = (const float*)d_in[13];
    const float* ln1_g    = (const float*)d_in[14];
    const float* ln1_b    = (const float*)d_in[15];
    const float* ln2_g    = (const float*)d_in[16];
    const float* ln2_b    = (const float*)d_in[17];

    char* ws = (char*)d_ws;
    size_t off = 0;
    auto alloc = [&](size_t bytes) -> char* {
        char* p = ws + off;
        off = (off + bytes + 255) & ~(size_t)255;
        return p;
    };
    float* seq      = (float*)alloc((size_t)NS * D * 4);
    float* l1       = (float*)alloc((size_t)NS * D * 4);
    float* r1       = (float*)alloc((size_t)NS * D * 4);
    float* h        = (float*)alloc((size_t)NS * D * 4);
    float* hid      = (float*)alloc((size_t)NS * 4 * D * 4);
    float* contents = (float*)alloc((size_t)NS * 4 * D * 4);
    u16*   winbf    = (u16*)  alloc((size_t)NS * 5 * D * 2);
    u16*   hidbf    = (u16*)  alloc((size_t)NS * 4 * D * 2);
    u16*   seqbf    = (u16*)  alloc((size_t)NS * D * 2);
    float* Lm       = (float*)alloc((size_t)NB * S * S * 4);
    float* Rm       = (float*)alloc((size_t)NB * S * S * 4);
    float* mask     = (float*)alloc((size_t)NS * 4);
    float* endm     = (float*)alloc((size_t)NS * 4);
    float* selm     = (float*)alloc((size_t)NS * 4);
    float* lastm    = (float*)alloc((size_t)NS * 4);
    float* ep       = (float*)alloc((size_t)NS * 4);
    float* sc       = (float*)alloc((size_t)NS * 4);
    float* tp       = (float*)alloc((size_t)NS * 4);
    float* um       = (float*)alloc((size_t)NB * 4);
    float* mx       = (float*)alloc(256);
    u16* WinitT = (u16*)alloc((size_t)1024 * 1024 * 2);
    u16* WcT    = (u16*)alloc((size_t)1024 * 5120 * 2);
    u16* W1T    = (u16*)alloc((size_t)4096 * 2048 * 2);
    u16* W2T    = (u16*)alloc((size_t)4096 * 4096 * 2);

    auto gemm = [&](const u16* A, int lda, const u16* Bt, int ldb,
                    float* C, int ldc, const float* bias, int M, int N, int K, int act) {
        int tilesM = M / 32, tilesN = N / 64;
        int T = tilesM * tilesN;
        int blocks = (T + 7) / 8;
        k_gemm<<<blocks, 256, 0, stream>>>(A, lda, Bt, ldb, C, ldc, bias,
                                           tilesM, tilesN, K, act);
    };

    // ---- weight prep (transpose + bf16) ----
    k_wt<<<(int)(((long long)1024 * 1024 + 255) / 256), 256, 0, stream>>>(W_init,  WinitT, 1024, 1024);
    k_wt<<<(int)(((long long)5120 * 1024 + 255) / 256), 256, 0, stream>>>(W_conv,  WcT,    5120, 1024);
    k_wt<<<(int)(((long long)2048 * 4096 + 255) / 256), 256, 0, stream>>>(W_cell1, W1T,    2048, 4096);
    k_wt<<<(int)(((long long)4096 * 4096 + 255) / 256), 256, 0, stream>>>(W_cell2, W2T,    4096, 4096);

    // ---- augment + init GEMM + LN1 ----
    k_augment<<<NS, 256, 0, stream>>>(in_seq, in_mask, STARTv, ENDv,
                                      seq, mask, endm, selm, lastm);
    k_cvtbf<<<NS, 256, 0, stream>>>(seq, D, seqbf, D, D, mask);   // (seq*mask) -> bf16
    gemm(seqbf, D, WinitT, D, h, D, b_init, NS, 1024, 1024, 0);
    k_ln1<<<NS, 256, 0, stream>>>(h, seq, ln1_g, ln1_b, mask);
    k_initstate<<<(NS + 255) / 256, 256, 0, stream>>>(mask, ep, um);

    // ---- 25-step scan ----
    for (int step = 0; step < STEPS; ++step) {
        k_neighbors<<<(2 * NS + 255) / 256, 256, 0, stream>>>(ep, mask, Lm, Rm);
        // win slices: [l2, l1, sq, r1, r2]
        k_rowmix<<<NS, 256, 0, stream>>>(Lm, seq, l1, winbf, 1 * D);          // l1
        k_rowmix<<<NS, 256, 0, stream>>>(Rm, seq, r1, winbf, 3 * D);          // r1
        k_rowmix<<<NS, 256, 0, stream>>>(Lm, l1, (float*)nullptr, winbf, 0);  // l2
        k_rowmix<<<NS, 256, 0, stream>>>(Rm, r1, (float*)nullptr, winbf, 4 * D); // r2
        k_cvtbf<<<NS, 256, 0, stream>>>(seq, D, winbf + 2 * D, 5 * D, D, (const float*)nullptr); // sq slice

        gemm(winbf, 5 * D, WcT, 5 * D, h, D, b_conv, NS, 1024, 5120, 1);      // h = gelu(win@Wc+b)

        k_zero1<<<1, 32, 0, stream>>>(mx);
        k_scorer1<<<NS / 8, 256, 0, stream>>>(h, W_scorer, b_scorer, sc, mx);
        k_scorer2<<<(NS + 255) / 256, 256, 0, stream>>>(sc, selm, um, mx, tp);

        // cat = win slices [l1, sq] : contiguous subview, lda = 5D
        gemm(winbf + 1 * D, 5 * D, W1T, 2 * D, hid, 4 * D, b_cell1, NS, 4096, 2048, 1);
        k_cvtbf<<<NS, 256, 0, stream>>>(hid, 4 * D, hidbf, 4 * D, 4 * D, (const float*)nullptr);
        gemm(hidbf, 4 * D, W2T, 4 * D, contents, 4 * D, b_cell2, NS, 4096, 4096, 0);

        k_comp<<<NS, 256, 0, stream>>>(contents, l1, seq, tp, ln2_g, ln2_b);
        k_ep<<<NB, 256, 0, stream>>>(Rm, tp, selm, ep, um);
    }

    // ---- outputs: seq[:,1:-1,:] then global_state ----
    float* out = (float*)d_out;
    k_outcopy<<<NB * SCT, 256, 0, stream>>>(seq, mask, out);
    k_outgs<<<NB, 256, 0, stream>>>(seq, mask, lastm, out + (size_t)NB * SCT * D);
}
